// SparseNeighConsensus_23545010717393
// MI455X (gfx1250) — matrix operand order repaired and checked by reading, not compiled, not run on hardware
//
#include <hip/hip_runtime.h>
#include <hip/hip_bf16.h>

// ---------------------------------------------------------------------------
// NCNet (4D neighborhood-consensus conv net) on MI455X via implicit-GEMM WMMA.
// Symmetric path handled by kernel-axis-swapped weights (no tensor transpose).
// Mixed precision: f16 operands, f32 accumulate, v_wmma_f32_16x16x32_f16.
// Halo staging via GLOBAL_LOAD_ASYNC_TO_LDS_B128 (ASYNCcnt path).
// ---------------------------------------------------------------------------

typedef __attribute__((ext_vector_type(16))) _Float16 v16h;
typedef __attribute__((ext_vector_type(8)))  _Float16 v8h;
typedef __attribute__((ext_vector_type(8)))  float    v8f;

#define SDIM 32
#define SPX  48          // padded LDS t-row: halo(-1) at idx 7, t=0..31 at 8..39, halo(+) at 40,41
#define DT   4           // d-tile per block
#define DTP  6           // DT + 2 halo

// Packed weight matrix offsets (in halves) inside wbuf (column-major [n][RK]):
//   per path: W1 @0 (16*96), W2 @1536 (16*832), W3 @14848 (16*832); path stride 28160
#define W2_OFF   1536
#define W3_OFF   14848
#define PATH_OFF 28160

// ---------------------------------------------------------------------------
// Pack weights into column-major f16 matrices [16][RK] so each lane's WMMA-B
// fragment is two contiguous 16B loads. r = kidx*CIN + ci; path 1 uses
// kernel-axis-swapped weights: w'(kh,kw,kd,kt) = w(kd,kt,kh,kw).
// ---------------------------------------------------------------------------
__global__ void prep_weights(const float* __restrict__ w1,
                             const float* __restrict__ w2,
                             const float* __restrict__ w3,
                             _Float16* __restrict__ wbuf) {
  int mat = blockIdx.x;              // 0..5
  int path = mat / 3, layer = mat % 3;
  int CIN  = (layer == 0) ? 1 : 10;
  int COUT = (layer == 2) ? 1 : 10;
  int RK   = (layer == 0) ? 96 : 832;
  int R    = CIN * 81;
  const float* w = (layer == 0) ? w1 : (layer == 1) ? w2 : w3;
  int base = path * PATH_OFF + ((layer == 0) ? 0 : (layer == 1) ? W2_OFF : W3_OFF);

  int total = RK * 16;
  for (int i = threadIdx.x; i < total; i += blockDim.x) {
    int r = i % RK;                  // column-major: i = n*RK + r
    int n = i / RK;
    float v = 0.0f;
    if (r < R && n < COUT) {
      int ci = r % CIN;
      int kidx = r / CIN;
      int q0 = kidx % 3, q1 = (kidx / 3) % 3, q2 = (kidx / 9) % 3, q3 = kidx / 27;
      int kh, kw, kd, kt;
      if (path == 0) { kh = q3; kw = q2; kd = q1; kt = q0; }
      else           { kh = q1; kw = q0; kd = q3; kt = q2; }  // swapped axes
      v = w[((((n * CIN + ci) * 3 + kh) * 3 + kw) * 3 + kd) * 3 + kt];
    }
    wbuf[base + i] = (_Float16)v;
  }
}

// Leading ReLU + f32 -> f16 conversion of the input tensor.
__global__ void relu_cvt(const float* __restrict__ x, _Float16* __restrict__ out, int n) {
  for (int i = blockIdx.x * blockDim.x + threadIdx.x; i < n; i += gridDim.x * blockDim.x)
    out[i] = (_Float16)fmaxf(x[i], 0.0f);
}

// ---------------------------------------------------------------------------
// Implicit-GEMM 4D conv layer. One block = one (b, h, w, d-tile of 4) pencil.
// M = 16 t-positions, N = 16 (cout padded), K = CIN*81 padded to 32-chunks.
// 8 waves; wave w owns (local d = w&3, t-half = w>>2) -> 1 WMMA per K-chunk.
// ---------------------------------------------------------------------------
template <int CIN, bool FINAL>
__global__ __launch_bounds__(256) void conv4d_wmma(
    const _Float16* __restrict__ actin,   // [B][CIN][32][32][32][32] f16
    const _Float16* __restrict__ gw,      // [16][RK] f16 packed weights (col-major)
    const float*    __restrict__ bias,    // [coutReal]
    _Float16*       __restrict__ actout,  // [B][coutReal][32^4] f16 (if !FINAL)
    float*          __restrict__ fout,    // [B][1][32^4] f32      (if FINAL)
    int coutReal, int accFlag)
{
  constexpr int R      = CIN * 81;
  constexpr int CHUNKS = (R + 31) / 32;
  constexpr int RK     = CHUNKS * 32;
  constexpr int XT     = CIN * 9 * DTP * SPX;  // halo tile halves
  constexpr int PADZ   = 192;                  // zero zone for padded K rows (max addr 182)
  constexpr int ZFILL  = (XT + PADZ) / 8;      // v8h zero-fill stores
  constexpr int SEGS   = CIN * 9 * DTP * 4;    // 16B async segments

  __shared__ __align__(16) _Float16 xtile[XT + PADZ];
  __shared__ int addrtab[RK];

  const int gid   = blockIdx.x;
  const int dtile = gid & 7;
  const int wq    = (gid >> 3) & 31;
  const int hq    = (gid >> 8) & 31;
  const int bq    = gid >> 13;
  const int tid   = threadIdx.x;
  const int d0    = dtile * DT;

  // --- Stage 1a: per-r LDS offset table + zero-prefill the tile -----------
  for (int r = tid; r < RK; r += 256) {
    int off;
    if (r < R) {
      int ci = r % CIN;
      int kidx = r / CIN;
      int kt = kidx % 3, kd = (kidx / 3) % 3, kw = (kidx / 9) % 3, kh = kidx / 27;
      off = (((ci * 3 + kh) * 3 + kw) * DTP + kd) * SPX + kt;
    } else {
      off = XT;                       // sentinel -> zeroed pad zone
    }
    addrtab[r] = off;
  }
  {
    v8h z = {};
    v8h* zp = (v8h*)xtile;
    for (int i = tid; i < ZFILL; i += 256) zp[i] = z;
  }
  __syncthreads();                    // zeros committed before async writes land

  // --- Stage 1b: async-copy valid 16B t-segments into LDS ------------------
  // Each task = one 16-byte chunk (8 halves, t in [8j, 8j+8)) of one
  // (ci,kh,kw,dl) row. h/w/d out-of-range rows stay zero; t halo stays zero.
  for (int task = tid; task < SEGS; task += 256) {
    int j = task & 3;
    int row = task >> 2;
    int dl = row % DTP;
    int r2 = row / DTP;
    int kw = r2 % 3; r2 /= 3;
    int kh = r2 % 3;
    int ci = r2 / 3;
    int hh = hq + kh - 1, ww = wq + kw - 1, dd = d0 + dl - 1;
    if ((unsigned)hh < SDIM && (unsigned)ww < SDIM && (unsigned)dd < SDIM) {
      size_t gidx = (((((size_t)bq * CIN + ci) * SDIM + hh) * SDIM + ww) * SDIM + dd) * SDIM
                  + 8 * j;                                  // 16B-aligned in global
      unsigned long long gaddr = (unsigned long long)(const void*)(actin + gidx);
      unsigned loff = (unsigned)(size_t)(const void*)&xtile[row * SPX + 8 + 8 * j]; // 16B-aligned LDS
      asm volatile("global_load_async_to_lds_b128 %0, %1, off"
                   :: "v"(loff), "v"(gaddr)
                   : "memory");
    }
  }
  asm volatile("s_wait_asynccnt 0x0" ::: "memory");
  __syncthreads();

  // --- Stage 2: WMMA main loop --------------------------------------------
  const int lane  = tid & 31;
  const int wave  = tid >> 5;
  const int dl    = wave & 3;        // local d
  const int t0    = (wave >> 2) << 4;// t-half: 0 or 16
  const int m     = lane & 15;       // A-matrix row = t offset (ISA A layout)
  const int grp   = lane >> 4;       // lane half selects K sub-range
  const int grp8  = grp << 3;
  const int grp16 = grp << 4;
  const int n     = lane & 15;       // B/C/D column = cout
  const int dbase = dl * SPX + t0 + m + 7;  // input t = out_t + kt - 1, stored at +8

  v8f c = {};

  #pragma unroll 1
  for (int ch = 0; ch < CHUNKS; ++ch) {
    const int rbase = ch << 5;

    // B fragment: two contiguous 16B loads from column-major weights (L2-hot)
    const v8h* bp = (const v8h*)(gw + (size_t)n * RK + rbase + grp16);
    v8h blo = bp[0], bhi = bp[1];
    v16h bfr;
    #pragma unroll
    for (int h = 0; h < 8; ++h) { bfr[h] = blo[h]; bfr[h + 8] = bhi[h]; }

    // A offsets for this chunk, then gather A fragment from LDS
    int offs[16];
    #pragma unroll
    for (int h = 0; h < 16; ++h) {
      int k = h + grp16;                    // contiguous K-order (matches B)
      offs[h] = addrtab[rbase + k];
    }
    v16h afr;
    #pragma unroll
    for (int h = 0; h < 16; ++h) afr[h] = xtile[offs[h] + dbase];

    c = __builtin_amdgcn_wmma_f32_16x16x32_f16(false, afr, false, bfr,
                                               (short)0, c, false, false);
  }

  // --- Stage 3: bias + ReLU epilogue --------------------------------------
  const float bn   = (n < coutReal) ? bias[n] : 0.0f;
  const int   moff = grp8;           // C/D rows: m = g + 8*grp
  const int   dgl  = d0 + dl;

  if constexpr (!FINAL) {
    if (n < coutReal) {
      size_t base = (((((size_t)bq * coutReal + n) * SDIM + hq) * SDIM + wq) * SDIM + dgl) * SDIM;
      v8h p;
      #pragma unroll
      for (int g = 0; g < 8; ++g) p[g] = (_Float16)fmaxf(c[g] + bn, 0.0f);
      *(v8h*)(actout + base + t0 + moff) = p;   // t-contiguous, 16B aligned
    }
  } else {
    if (n == 0) {                     // lanes 0 (rows 0-7) and 16 (rows 8-15)
      size_t base = ((((size_t)bq * SDIM + hq) * SDIM + wq) * SDIM + dgl) * SDIM;
      #pragma unroll
      for (int g = 0; g < 8; ++g) {
        float y = fmaxf(c[g] + bn, 0.0f);
        size_t i0 = base + t0 + moff + g;
        if (accFlag) fout[i0] += y;
        else         fout[i0]  = y;
      }
    }
  }
}

// ---------------------------------------------------------------------------
extern "C" void kernel_launch(void* const* d_in, const int* in_sizes, int n_in,
                              void* d_out, int out_size, void* d_ws, size_t ws_size,
                              hipStream_t stream) {
  const float* x  = (const float*)d_in[0];
  const float* w1 = (const float*)d_in[1];
  const float* b1 = (const float*)d_in[2];
  const float* w2 = (const float*)d_in[3];
  const float* b2 = (const float*)d_in[4];
  const float* w3 = (const float*)d_in[5];
  const float* b3 = (const float*)d_in[6];

  char* ws = (char*)d_ws;
  _Float16* wbuf = (_Float16*)ws;                                     // 128 KB
  _Float16* act0 = (_Float16*)(ws + (size_t)(128 << 10));             // 4 MB  (1 ch)
  _Float16* act1 = (_Float16*)(ws + (size_t)(128 << 10) + (4u << 20));          // 40 MB (10 ch)
  _Float16* act2 = (_Float16*)(ws + (size_t)(128 << 10) + (4u << 20) + (40ull << 20)); // 40 MB
  float* out = (float*)d_out;

  const int N0 = 2 * 1 * SDIM * SDIM * SDIM * SDIM;  // 2,097,152
  const int NBLK = 2 * SDIM * SDIM * (SDIM / DT);    // 16384 pencils

  prep_weights<<<6, 256, 0, stream>>>(w1, w2, w3, wbuf);
  relu_cvt<<<4096, 256, 0, stream>>>(x, act0, N0);

  for (int p = 0; p < 2; ++p) {
    const _Float16* wp = wbuf + (size_t)p * PATH_OFF;
    conv4d_wmma<1,  false><<<NBLK, 256, 0, stream>>>(act0, wp,          b1, act1, nullptr, 10, 0);
    conv4d_wmma<10, false><<<NBLK, 256, 0, stream>>>(act1, wp + W2_OFF, b2, act2, nullptr, 10, 0);
    conv4d_wmma<10, true ><<<NBLK, 256, 0, stream>>>(act2, wp + W3_OFF, b3, nullptr, out,   1, p);
  }
}
